// ConsistencyLoss_37254546325509
// MI455X (gfx1250) — matrix operand order repaired and checked by reading, not compiled, not run on hardware
//
#include <hip/hip_runtime.h>
#include <hip/hip_bf16.h>

typedef __attribute__((ext_vector_type(16))) __bf16 v16bf;
typedef __attribute__((ext_vector_type(8)))  __bf16 v8bf;
typedef __attribute__((ext_vector_type(8)))  float  v8f;

#define BATCH 4
#define CDIM  128
#define NDIM  4096        // 64*64
#define TN    4           // N-tiles per wave (macro-tile 16x64)

// ---------------------------------------------------------------------------
// Prep: per (b,n) column of src[b,c,n] (fp32, c-stride = N), normalize by its
// L2 norm over c, convert to bf16, store transposed dst[b,n,c] (c contiguous).
// ---------------------------------------------------------------------------
__global__ void cl_prep_normalize_bf16(const float* __restrict__ src,
                                       __bf16* __restrict__ dst) {
    int idx = blockIdx.x * blockDim.x + threadIdx.x;   // over BATCH*NDIM
    int b = idx >> 12;            // idx / NDIM
    int n = idx & (NDIM - 1);     // idx % NDIM
    const float* col = src + ((size_t)b * CDIM) * NDIM + n;
    float ss = 0.f;
    #pragma unroll 8
    for (int c = 0; c < CDIM; ++c) {
        float v = col[(size_t)c * NDIM];
        ss += v * v;
    }
    float r = (ss > 0.f) ? rsqrtf(ss) : 0.f;
    __bf16* out = dst + (size_t)idx * CDIM;
    #pragma unroll 8
    for (int c = 0; c < CDIM; ++c) {
        out[c] = (__bf16)(col[(size_t)c * NDIM] * r);
    }
}

__device__ __forceinline__ v16bf cl_cat(v8bf lo, v8bf hi) {
    return __builtin_shufflevector(lo, hi, 0, 1, 2, 3, 4, 5, 6, 7,
                                           8, 9, 10, 11, 12, 13, 14, 15);
}

// A fragment (16x32 bf16, M x K): lane = (h = lane>>4, m = lane&15)
// V0-3 = K[k0+8h .. k0+8h+7], V4-7 = K[k0+16+8h .. +7]  -> two b128 loads
__device__ __forceinline__ v16bf cl_ldfrag_a(const __bf16* __restrict__ p,
                                             int k0, int h) {
    v8bf lo = *(const v8bf*)(p + k0 + 8 * h);
    v8bf hi = *(const v8bf*)(p + k0 + 16 + 8 * h);
    return cl_cat(lo, hi);
}

// B fragment (32x16 bf16, K x N): lane = (h, n); V0-3 = K[k0+8h .. +7],
// V4-7 = K[k0+16+8h .. +7] -> two b128 loads (same order as A)
__device__ __forceinline__ v16bf cl_ldfrag_b(const __bf16* __restrict__ p,
                                             int k0, int h) {
    v8bf lo = *(const v8bf*)(p + k0 + 8 * h);
    v8bf hi = *(const v8bf*)(p + k0 + 16 + 8 * h);
    return cl_cat(lo, hi);
}

// ---------------------------------------------------------------------------
// Fused bf16-WMMA GEMM (dots = momT x base, both pre-normalized) + mask
// select + sum/count reduction. One wave per 16x64 output macro-tile.
// ---------------------------------------------------------------------------
__global__ void cl_gemm_mask_reduce(const __bf16* __restrict__ nmomT,
                                    const __bf16* __restrict__ nbaseT,
                                    const int* __restrict__ matrix,
                                    float* __restrict__ gsum,
                                    unsigned* __restrict__ gcnt) {
    const int lane = threadIdx.x & 31;
    const int wib  = threadIdx.x >> 5;                       // wave in block
    const int wave = blockIdx.x * (blockDim.x >> 5) + wib;   // global wave id

    const int TM_TILES = NDIM / 16;        // 256
    const int NG_TILES = NDIM / (16 * TN); // 64
    const int b  = wave / (TM_TILES * NG_TILES);
    const int r0 = wave % (TM_TILES * NG_TILES);
    const int tm = r0 / NG_TILES;          // M tile index (16 rows)
    const int tg = r0 % NG_TILES;          // N macro-tile index (64 cols)

    const int h = lane >> 4;
    const int l = lane & 15;

    const int mrow = tm * 16 + l;          // A row for this lane
    const int ncol0 = tg * (16 * TN);      // first output column of macro-tile

    const __bf16* aptr = nmomT + ((size_t)b * NDIM + mrow) * CDIM;
    const __bf16* bptr[TN];
    #pragma unroll
    for (int j = 0; j < TN; ++j)
        bptr[j] = nbaseT + ((size_t)b * NDIM + ncol0 + j * 16 + l) * CDIM;

    // Prefetch the mask tile rows we will need in the epilogue.
    const int* mrow_ptr =
        matrix + ((size_t)b * NDIM + tm * 16 + 8 * h) * NDIM + ncol0 + l;
    __builtin_prefetch(mrow_ptr, 0, 1);

    v8f acc[TN];
    #pragma unroll
    for (int j = 0; j < TN; ++j) acc[j] = (v8f){0.f, 0.f, 0.f, 0.f,
                                                0.f, 0.f, 0.f, 0.f};

    #pragma unroll
    for (int ks = 0; ks < CDIM / 32; ++ks) {   // 4 k-steps
        const int k0 = ks * 32;
        v16bf afrag = cl_ldfrag_a(aptr, k0, h);
        #pragma unroll
        for (int j = 0; j < TN; ++j) {
            v16bf bfrag = cl_ldfrag_b(bptr[j], k0, h);
            acc[j] = __builtin_amdgcn_wmma_f32_16x16x32_bf16(
                false, afrag, false, bfrag, (short)0, acc[j], false, false);
        }
    }

    // Epilogue: mask-select and accumulate. C/D layout: VGPR r, lanes 0-15 ->
    // (M=r, N=lane); lanes 16-31 -> (M=r+8, N=lane-16).
    float sum = 0.f;
    unsigned cnt = 0u;
    #pragma unroll
    for (int j = 0; j < TN; ++j) {
        const size_t mbase =
            ((size_t)b * NDIM + tm * 16 + 8 * h) * NDIM + ncol0 + j * 16 + l;
        #pragma unroll
        for (int r = 0; r < 8; ++r) {
            int mk = matrix[mbase + (size_t)r * NDIM];
            sum += (mk != 0) ? acc[j][r] : 0.f;
            cnt += (mk != 0) ? 1u : 0u;
        }
    }

    // Wave reduction (wave32)
    #pragma unroll
    for (int off = 16; off > 0; off >>= 1) {
        sum += __shfl_down(sum, off, 32);
        cnt += (unsigned)__shfl_down((int)cnt, off, 32);
    }

    __shared__ float    lsum[8];
    __shared__ unsigned lcnt[8];
    if (lane == 0) { lsum[wib] = sum; lcnt[wib] = cnt; }
    __syncthreads();
    if (threadIdx.x == 0) {
        float    ts = 0.f;
        unsigned tc = 0u;
        #pragma unroll
        for (int i = 0; i < 8; ++i) { ts += lsum[i]; tc += lcnt[i]; }
        atomicAdd(gsum, ts);
        atomicAdd(gcnt, tc);
    }
}

__global__ void cl_finalize(const float* __restrict__ gsum,
                            const unsigned* __restrict__ gcnt,
                            float* __restrict__ out) {
    if (threadIdx.x == 0 && blockIdx.x == 0) {
        float c = (float)(*gcnt);
        out[0] = (c > 0.f) ? (-(*gsum) / c) : 0.f;
    }
}

// ---------------------------------------------------------------------------
extern "C" void kernel_launch(void* const* d_in, const int* in_sizes, int n_in,
                              void* d_out, int out_size, void* d_ws, size_t ws_size,
                              hipStream_t stream) {
    const float* en_base = (const float*)d_in[0];   // [B, C, H, W] fp32
    const float* en_mom  = (const float*)d_in[1];   // [B, C, H, W] fp32
    const int*   matrix  = (const int*)d_in[2];     // [B, N, N] int32
    float* out = (float*)d_out;

    const size_t BNC = (size_t)BATCH * NDIM * CDIM;           // 2 Mi elements
    char* ws = (char*)d_ws;
    __bf16*   nmomT  = (__bf16*)ws;                           // 4 MB
    __bf16*   nbaseT = (__bf16*)(ws + BNC * sizeof(__bf16));  // 4 MB
    float*    gsum   = (float*)(ws + 2 * BNC * sizeof(__bf16));
    unsigned* gcnt   = (unsigned*)(ws + 2 * BNC * sizeof(__bf16) + sizeof(float));

    hipMemsetAsync(ws + 2 * BNC * sizeof(__bf16), 0,
                   sizeof(float) + sizeof(unsigned), stream);

    // Normalize + bf16 + transpose: one thread per (b, n) column.
    {
        const int threads = BATCH * NDIM;   // 16384
        cl_prep_normalize_bf16<<<threads / 256, 256, 0, stream>>>(en_mom,  nmomT);
        cl_prep_normalize_bf16<<<threads / 256, 256, 0, stream>>>(en_base, nbaseT);
    }

    // GEMM + mask + reduce: one wave per 16x64 macro-tile.
    {
        const int waves  = BATCH * (NDIM / 16) * (NDIM / (16 * TN)); // 65536
        const int blocks = waves / 8;                                // 8 waves/block
        cl_gemm_mask_reduce<<<blocks, 256, 0, stream>>>(nmomT, nbaseT, matrix,
                                                        gsum, gcnt);
    }

    cl_finalize<<<1, 32, 0, stream>>>(gsum, gcnt, out);
}
